// GAT_46239617908905
// MI455X (gfx1250) — matrix operand order repaired and checked by reading, not compiled, not run on hardware
//
#include <hip/hip_runtime.h>
#include <hip/hip_bf16.h>

// GAT (3-layer) for MI455X / gfx1250.
// GEMMs: v_wmma_f32_16x16x32_bf16, 16x64 output strip per wave. K=128 fully
// unrolled: all A/B fragments loaded up-front (one big b128 clause), a
// sched_barrier pins the load phase, then 16 WMMAs retire against staggered
// partial s_wait_loadcnt. Edge softmax + aggregation: memory/atomic bound.

#define N_NODES 20000
#define N_EDGES 200000
#define DM      128
#define NH      8
#define KOUT_LAST 1024   // NH * DM

typedef __attribute__((ext_vector_type(16))) __bf16 v16bf;
typedef __attribute__((ext_vector_type(8)))  float  v8f;

// ---------- helpers ----------
__device__ __forceinline__ unsigned int f2bf_u(float f) {
  unsigned int u = __float_as_uint(f);
  u += 0x7fffu + ((u >> 16) & 1u);           // round-to-nearest-even
  return u >> 16;
}
// monotonic float <-> int key for atomicMax-based segment max
__device__ __forceinline__ int f32_key(float f) {
  int i = __float_as_int(f);
  return i >= 0 ? i : (i ^ 0x7fffffff);
}
__device__ __forceinline__ float key_f32(int i) {
  return __int_as_float(i >= 0 ? i : (i ^ 0x7fffffff));
}

// ---------- casts ----------
// 4 floats -> 4 bf16 per thread (float4 in, packed uint2 out)
__global__ void k_cast_bf16_v4(const float* __restrict__ in,
                               unsigned short* __restrict__ out, int n4) {
  int i = blockIdx.x * blockDim.x + threadIdx.x;
  if (i >= n4) return;
  float4 f = ((const float4*)in)[i];
  uint2 p;
  p.x = f2bf_u(f.x) | (f2bf_u(f.y) << 16);
  p.y = f2bf_u(f.z) | (f2bf_u(f.w) << 16);
  ((uint2*)out)[i] = p;
}

// W: [128][kout] row-major -> Wt: [kout][128] bf16 (K-contiguous per out col)
__global__ void k_cast_wt_bf16(const float* __restrict__ W,
                               unsigned short* __restrict__ Wt, int kout) {
  int i = blockIdx.x * blockDim.x + threadIdx.x;
  if (i < kout * DM) {
    int n = i / DM, k = i % DM;
    Wt[i] = (unsigned short)f2bf_u(W[k * kout + n]);
  }
}

// ---------- bf16 WMMA GEMM: C[N_NODES x kout] = A[N_NODES x 128] * W ----------
// One wave per 16x64 output strip. Load phase (one big clause) then 16 WMMAs.
__global__ void __launch_bounds__(64)
k_gemm_bf16(const unsigned short* __restrict__ A,   // [N][128] bf16
            const unsigned short* __restrict__ Bt,  // [kout][128] bf16 (W^T)
            float* __restrict__ C, int kout) {
  const int lane = threadIdx.x;           // 0..31 (wave32)
  const int l15  = lane & 15;
  const int hi   = lane >> 4;
  const int m0 = blockIdx.x * 16;
  const int n0 = (blockIdx.y * 2 + threadIdx.y) * 64;  // 64-col strip per wave

  union Frag { uint4 q[2]; v16bf v; };
  Frag a[4];        // 4 k-steps of A (row fragment, reused across 4 N-tiles)
  Frag b[4][4];     // [k-step][N-tile]
  v8f  c[4] = {{}, {}, {}, {}};

  // A lane layout: row = l15; K = 8*hi + {0..7} and 8*hi + 16 + {0..7}
  const unsigned short* arow = A  + (size_t)(m0 + l15) * DM + hi * 8;
  // B lane layout: col = l15; K = 8*hi + {0..7} and 8*hi + 16 + {0..7}
  const unsigned short* bcol = Bt + (size_t)(n0 + l15) * DM + hi * 8;
  const int TS = 16 * DM;                 // elements between adjacent N-tiles of Wt

  // ---- load phase: 8 + 32 b128 loads, issue order matches consume order ----
#pragma unroll
  for (int kk = 0; kk < 4; ++kk) {
    const int k0 = kk * 32;
    a[kk].q[0] = *(const uint4*)(arow + k0);
    a[kk].q[1] = *(const uint4*)(arow + k0 + 16);
#pragma unroll
    for (int t = 0; t < 4; ++t) {
      b[kk][t].q[0] = *(const uint4*)(bcol + t * TS + k0);
      b[kk][t].q[1] = *(const uint4*)(bcol + t * TS + k0 + 16);
    }
  }

  // Pin the load phase: nothing may be scheduled across this point, so the
  // scheduler cannot sink loads down next to their WMMAs (which would force
  // full s_wait_loadcnt 0x0 drains). WMMAs below then use partial waits.
  __builtin_amdgcn_sched_barrier(0);

  // ---- compute phase: 16 WMMAs, earliest-loaded operands consumed first ----
#pragma unroll
  for (int kk = 0; kk < 4; ++kk)
#pragma unroll
    for (int t = 0; t < 4; ++t)
      c[t] = __builtin_amdgcn_wmma_f32_16x16x32_bf16(
          false, a[kk].v, false, b[kk][t].v, (short)0, c[t], false, false);

  // D layout: VGPR v -> row m0 + v + 8*hi, col tile*16 + n0 + l15
  float* crow = C + (size_t)(m0 + hi * 8) * kout + n0 + l15;
#pragma unroll
  for (int v = 0; v < 8; ++v) {
#pragma unroll
    for (int t = 0; t < 4; ++t)
      crow[(size_t)v * kout + t * 16] = c[t][v];
  }
}

// ---------- attention scores: el/er[n,h] = <feat[n,h,:], al/ar[h,:]> ----------
__global__ void k_attn(const float* __restrict__ feat, const float* __restrict__ al,
                       const float* __restrict__ ar, float* __restrict__ el,
                       float* __restrict__ er, int dh) {
  int i = blockIdx.x * blockDim.x + threadIdx.x;   // i = n*NH + h
  if (i >= N_NODES * NH) return;
  int hh = i % NH;
  const float* f   = feat + (size_t)i * dh;
  const float* pal = al + hh * dh;
  const float* par = ar + hh * dh;
  float s1 = 0.f, s2 = 0.f;
  for (int d = 0; d < dh; ++d) { float x = f[d]; s1 += x * pal[d]; s2 += x * par[d]; }
  el[i] = s1; er[i] = s2;
}

// ---------- fills (b128 stores) ----------
__global__ void k_fill4_u32(uint4* __restrict__ p, unsigned int v, size_t n4) {
  size_t i = (size_t)blockIdx.x * blockDim.x + threadIdx.x;
  if (i < n4) { uint4 q = {v, v, v, v}; p[i] = q; }
}

// ---------- edge pipeline ----------
__global__ void k_edge_logit_max(const int* __restrict__ src, const int* __restrict__ dst,
                                 const float* __restrict__ el, const float* __restrict__ er,
                                 float* __restrict__ coef, int* __restrict__ mkey) {
  int i = blockIdx.x * blockDim.x + threadIdx.x;   // i = e*NH + h
  if (i >= N_EDGES * NH) return;
  int e = i >> 3, hh = i & 7;                      // NH == 8
  float v = el[src[e] * NH + hh] + er[dst[e] * NH + hh];
  v = v > 0.f ? v : 0.2f * v;                      // leaky_relu(0.2)
  coef[i] = v;
  atomicMax(&mkey[dst[e] * NH + hh], f32_key(v));
}

__global__ void k_edge_exp_sum(const int* __restrict__ dst, const int* __restrict__ mkey,
                               float* __restrict__ coef, float* __restrict__ z) {
  int i = blockIdx.x * blockDim.x + threadIdx.x;
  if (i >= N_EDGES * NH) return;
  int e = i >> 3, hh = i & 7;
  int t = dst[e] * NH + hh;
  float p = __expf(coef[i] - key_f32(mkey[t]));
  coef[i] = p;
  atomicAdd(&z[t], p);
}

__global__ void k_edge_norm(const int* __restrict__ dst, const float* __restrict__ z,
                            float* __restrict__ coef) {
  int i = blockIdx.x * blockDim.x + threadIdx.x;
  if (i >= N_EDGES * NH) return;
  int e = i >> 3, hh = i & 7;
  coef[i] = coef[i] / z[dst[e] * NH + hh];
}

// 2D grid: (edge, channel-chunk). src/dst block-uniform -> scalar loads.
__global__ void __launch_bounds__(128)
k_edge_aggr(const int* __restrict__ src, const int* __restrict__ dst,
            const float* __restrict__ feat, const float* __restrict__ coef,
            float* __restrict__ rst, int dtot, int dhShift) {
  int e = blockIdx.x;
  int c = blockIdx.y * 128 + threadIdx.x;
  int s = src[e], t = dst[e];
  float a = coef[e * NH + (c >> dhShift)];
  atomicAdd(&rst[(size_t)t * dtot + c], feat[(size_t)s * dtot + c] * a);
}

// ---------- epilogues ----------
__global__ void k_finish_concat(const float* __restrict__ hold, const float* __restrict__ rst,
                                const float* __restrict__ b, float* __restrict__ hnew) {
  int i = blockIdx.x * blockDim.x + threadIdx.x;
  if (i >= N_NODES * DM) return;
  float t = rst[i] + b[i & (DM - 1)];
  float e = t > 0.f ? t : (__expf(t) - 1.f);       // ELU
  hnew[i] = hold[i] + e;
}

__global__ void k_finish_mean(const float* __restrict__ h, const float* __restrict__ rst,
                              const float* __restrict__ bl, float* __restrict__ out) {
  int i = blockIdx.x * blockDim.x + threadIdx.x;
  if (i >= N_NODES * DM) return;
  int n = i >> 7, c = i & 127;
  float s = 0.f;
#pragma unroll
  for (int hh = 0; hh < NH; ++hh)
    s += rst[(size_t)n * KOUT_LAST + hh * DM + c] + bl[hh * DM + c];
  out[i] = h[i] + s * 0.125f;                      // mean over 8 heads + residual
}

// ---------- host-side layer driver ----------
struct WsPtrs {
  float* h_ws; unsigned short* hbf; unsigned short* wtbf; float* feat;
  float* el; float* er; int* mkey; float* z; float* coef; float* rst;
};

static void run_gat_common(const float* hin, const float* W, const float* al,
                           const float* ar, int kout, const int* src, const int* dst,
                           const WsPtrs& w, hipStream_t s) {
  const int dh = kout / NH;                  // 16 or 128
  const int dhShift = (dh == 16) ? 4 : 7;
  const int B = 256;
  auto gb = [](long long n, int b) { return dim3((unsigned)((n + b - 1) / b)); };

  k_cast_bf16_v4<<<gb((long long)N_NODES * DM / 4, B), B, 0, s>>>(hin, w.hbf, N_NODES * DM / 4);
  k_cast_wt_bf16<<<gb((long long)kout * DM, B), B, 0, s>>>(W, w.wtbf, kout);

  dim3 gg(N_NODES / 16, kout / 128), bg(32, 2);  // one wave per 16x64 strip
  k_gemm_bf16<<<gg, bg, 0, s>>>(w.hbf, w.wtbf, w.feat, kout);

  k_attn<<<gb((long long)N_NODES * NH, B), B, 0, s>>>(w.feat, al, ar, w.el, w.er, dh);

  k_fill4_u32<<<gb((long long)N_NODES * NH / 4, B), B, 0, s>>>(
      (uint4*)w.mkey, 0x80000000u, (size_t)N_NODES * NH / 4);
  k_fill4_u32<<<gb((long long)N_NODES * NH / 4, B), B, 0, s>>>(
      (uint4*)w.z, 0u, (size_t)N_NODES * NH / 4);
  k_fill4_u32<<<gb((long long)N_NODES * kout / 4, B), B, 0, s>>>(
      (uint4*)w.rst, 0u, (size_t)N_NODES * kout / 4);

  k_edge_logit_max<<<gb((long long)N_EDGES * NH, B), B, 0, s>>>(src, dst, w.el, w.er, w.coef, w.mkey);
  k_edge_exp_sum<<<gb((long long)N_EDGES * NH, B), B, 0, s>>>(dst, w.mkey, w.coef, w.z);
  k_edge_norm<<<gb((long long)N_EDGES * NH, B), B, 0, s>>>(dst, w.z, w.coef);

  dim3 ga(N_EDGES, kout / 128);
  k_edge_aggr<<<ga, 128, 0, s>>>(src, dst, w.feat, w.coef, w.rst, kout, dhShift);
}

extern "C" void kernel_launch(void* const* d_in, const int* in_sizes, int n_in,
                              void* d_out, int out_size, void* d_ws, size_t ws_size,
                              hipStream_t stream) {
  const float* x       = (const float*)d_in[0];
  const int*   src     = (const int*)d_in[1];
  const int*   dst     = (const int*)d_in[2];
  const float* Ws      = (const float*)d_in[3];   // [2][128][128]
  const float* als     = (const float*)d_in[4];   // [2][8][16]
  const float* ars     = (const float*)d_in[5];
  const float* bs      = (const float*)d_in[6];   // [2][128]
  const float* W_last  = (const float*)d_in[7];   // [128][1024]
  const float* al_last = (const float*)d_in[8];   // [8][128]
  const float* ar_last = (const float*)d_in[9];
  const float* b_last  = (const float*)d_in[10];  // [1024]

  // workspace carve-out (~189 MB total)
  size_t off = 0;
  auto take = [&](size_t bytes) {
    void* p = (char*)d_ws + off;
    off += (bytes + 255) & ~(size_t)255;
    return p;
  };
  WsPtrs w;
  w.h_ws = (float*)take((size_t)N_NODES * DM * 4);
  w.hbf  = (unsigned short*)take((size_t)N_NODES * DM * 2);
  w.wtbf = (unsigned short*)take((size_t)KOUT_LAST * DM * 2);
  w.feat = (float*)take((size_t)N_NODES * KOUT_LAST * 4);
  w.el   = (float*)take((size_t)N_NODES * NH * 4);
  w.er   = (float*)take((size_t)N_NODES * NH * 4);
  w.mkey = (int*)take((size_t)N_NODES * NH * 4);
  w.z    = (float*)take((size_t)N_NODES * NH * 4);
  w.coef = (float*)take((size_t)N_EDGES * NH * 4);
  w.rst  = (float*)take((size_t)N_NODES * KOUT_LAST * 4);

  const int B = 256;
  auto gb = [](long long n, int b) { return dim3((unsigned)((n + b - 1) / b)); };

  // layer 0: hin = x, output -> h_ws
  run_gat_common(x, Ws, als, ars, DM, src, dst, w, stream);
  k_finish_concat<<<gb((long long)N_NODES * DM, B), B, 0, stream>>>(x, w.rst, bs, w.h_ws);

  // layer 1: hin = h_ws, output -> h_ws (elementwise in-place is safe)
  run_gat_common(w.h_ws, Ws + DM * DM, als + NH * 16, ars + NH * 16, DM, src, dst, w, stream);
  k_finish_concat<<<gb((long long)N_NODES * DM, B), B, 0, stream>>>(w.h_ws, w.rst, bs + DM, w.h_ws);

  // last layer: kout = 1024, mean over heads + residual -> d_out
  run_gat_common(w.h_ws, W_last, al_last, ar_last, KOUT_LAST, src, dst, w, stream);
  k_finish_mean<<<gb((long long)N_NODES * DM, B), B, 0, stream>>>(w.h_ws, w.rst, b_last, (float*)d_out);
}
